// PointNet2SSGSegmentation_8452495638618
// MI455X (gfx1250) — matrix operand order repaired and checked by reading, not compiled, not run on hardware
//
#include <hip/hip_runtime.h>
#include <hip/hip_bf16.h>

// PointNet++ SSG segmentation forward, CDNA5 (gfx1250), wave32.
// Shared-MLP 1x1convs + classifier run on v_wmma_f32_16x16x32_bf16.
// Block = 2 waves = 64(o) x 16(n) macro tile. The 32K x 16N activation tile
// is staged in LDS once per K-step via global_load_async_to_lds_b128,
// shared by both waves; B fragments then come from LDS with immediate
// offsets instead of 16 dynamically-strided global pointers.
// SA-stage max-pool fused into the WMMA epilogue via float-bits atomicMax.

typedef __attribute__((ext_vector_type(16))) __bf16 v16bf;
typedef __attribute__((ext_vector_type(8)))  float  v8f;
typedef int v4i __attribute__((vector_size(16)));   // matches builtin param

#define BN_EPS 1e-5f

#if defined(__has_builtin)
#if __has_builtin(__builtin_amdgcn_global_load_async_to_lds_b128) && \
    __has_builtin(__builtin_amdgcn_s_wait_asynccnt)
#define HAVE_ASYNC_LDS 1
#endif
#endif
#ifndef HAVE_ASYNC_LDS
#define HAVE_ASYNC_LDS 0
#endif

// ---------------------------------------------------------------- utilities
__global__ void k_fill_zero(float* __restrict__ p, int n) {
  int i = blockIdx.x * blockDim.x + threadIdx.x;
  if (i < n) p[i] = 0.f;
}

// x (B,N,9) -> xyz (B,N,3), feat (B,6,N)
__global__ void k_split(const float* __restrict__ x, float* __restrict__ xyz,
                        float* __restrict__ feat, int B_, int N_, int F_) {
  int i = blockIdx.x * blockDim.x + threadIdx.x;
  if (i >= B_ * N_) return;
  int b = i / N_, n = i % N_;
  const float* src = x + (size_t)i * F_;
  xyz[(size_t)i * 3 + 0] = src[0];
  xyz[(size_t)i * 3 + 1] = src[1];
  xyz[(size_t)i * 3 + 2] = src[2];
  for (int c = 0; c < F_ - 3; ++c)
    feat[((size_t)b * (F_ - 3) + c) * N_ + n] = src[3 + c];
}

// ------------------------------------------------- farthest point sampling
__global__ void k_fps(const float* __restrict__ xyz, int* __restrict__ fidx,
                      int N_, int np) {
  extern __shared__ char smem[];
  float* dist = (float*)smem;                // N_ floats
  float* rv   = dist + N_;                   // blockDim floats
  int*   ri   = (int*)(rv + blockDim.x);     // blockDim ints
  __shared__ int s_last;
  int b = blockIdx.x;
  const float* X = xyz + (size_t)b * N_ * 3;
  for (int i = threadIdx.x; i < N_; i += blockDim.x) dist[i] = 1e10f;
  if (threadIdx.x == 0) { s_last = 0; fidx[b * np] = 0; }
  __syncthreads();
  for (int it = 1; it < np; ++it) {
    int last = s_last;
    float px = X[last * 3], py = X[last * 3 + 1], pz = X[last * 3 + 2];
    float bestv = -1.f; int besti = 0;
    for (int i = threadIdx.x; i < N_; i += blockDim.x) {
      float dx = X[i * 3] - px, dy = X[i * 3 + 1] - py, dz = X[i * 3 + 2] - pz;
      float d = dx * dx + dy * dy + dz * dz;
      float dm = fminf(dist[i], d);
      dist[i] = dm;
      if (dm > bestv) { bestv = dm; besti = i; }
    }
    rv[threadIdx.x] = bestv; ri[threadIdx.x] = besti;
    __syncthreads();
    for (int s = blockDim.x / 2; s > 0; s >>= 1) {
      if (threadIdx.x < s) {
        if (rv[threadIdx.x + s] > rv[threadIdx.x] ||
            (rv[threadIdx.x + s] == rv[threadIdx.x] &&
             ri[threadIdx.x + s] < ri[threadIdx.x])) {
          rv[threadIdx.x] = rv[threadIdx.x + s];
          ri[threadIdx.x] = ri[threadIdx.x + s];
        }
      }
      __syncthreads();
    }
    if (threadIdx.x == 0) { s_last = ri[0]; fidx[b * np + it] = ri[0]; }
    __syncthreads();
  }
}

__global__ void k_gather_xyz(const float* __restrict__ xyz,
                             const int* __restrict__ fidx,
                             float* __restrict__ nxyz, int B_, int N_, int np) {
  int i = blockIdx.x * blockDim.x + threadIdx.x;
  if (i >= B_ * np) return;
  int b = i / np;
  int idx = fidx[i];
  const float* P = xyz + ((size_t)b * N_ + idx) * 3;
  nxyz[(size_t)i * 3 + 0] = P[0];
  nxyz[(size_t)i * 3 + 1] = P[1];
  nxyz[(size_t)i * 3 + 2] = P[2];
}

// Sequential scan == reference's top_k-of-smallest-indices-within-radius.
__global__ void k_ballquery(const float* __restrict__ xyz,
                            const float* __restrict__ nxyz,
                            int* __restrict__ gidx, int B_, int N_, int np,
                            int ns, float r2) {
  int i = blockIdx.x * blockDim.x + threadIdx.x;
  if (i >= B_ * np) return;
  int b = i / np;
  const float* X = xyz + (size_t)b * N_ * 3;
  float cx = nxyz[(size_t)i * 3], cy = nxyz[(size_t)i * 3 + 1],
        cz = nxyz[(size_t)i * 3 + 2];
  int* out = gidx + (size_t)i * ns;
  int cnt = 0, first = 0;
  for (int j = 0; j < N_ && cnt < ns; ++j) {
    float dx = X[j * 3] - cx, dy = X[j * 3 + 1] - cy, dz = X[j * 3 + 2] - cz;
    if (dx * dx + dy * dy + dz * dz <= r2) {
      if (cnt == 0) first = j;
      out[cnt++] = j;
    }
  }
  for (int s = cnt; s < ns; ++s) out[s] = first;
}

// grouped (B, 3+Cf, np*ns): centered xyz then gathered feats
__global__ void k_group(const float* __restrict__ xyz,
                        const float* __restrict__ nxyz,
                        const float* __restrict__ feats,
                        const int* __restrict__ gidx, float* __restrict__ grouped,
                        int B_, int N_, int np, int ns, int Cf) {
  int i = blockIdx.x * blockDim.x + threadIdx.x;  // over B*np*ns
  if (i >= B_ * np * ns) return;
  int b = i / (np * ns);
  int rem = i % (np * ns);
  int p = rem / ns;
  int idx = gidx[i];
  const float* P = xyz + ((size_t)b * N_ + idx) * 3;
  const float* Cn = nxyz + ((size_t)b * np + p) * 3;
  size_t M = (size_t)np * ns;
  float* G = grouped + (size_t)b * (3 + Cf) * M;
  G[0 * M + rem] = P[0] - Cn[0];
  G[1 * M + rem] = P[1] - Cn[1];
  G[2 * M + rem] = P[2] - Cn[2];
  const float* F = feats + (size_t)b * Cf * N_;
  for (int c = 0; c < Cf; ++c) G[(size_t)(3 + c) * M + rem] = F[(size_t)c * N_ + idx];
}

// SA4 global grouping: (B, 3+Cf, n) = concat(xyz^T, feats) (no centering)
__global__ void k_sa4_group(const float* __restrict__ xyz,
                            const float* __restrict__ feats,
                            float* __restrict__ grouped, int B_, int n, int Cf) {
  int Cin = 3 + Cf;
  int i = blockIdx.x * blockDim.x + threadIdx.x;
  if (i >= B_ * Cin * n) return;
  int b = i / (Cin * n);
  int rem = i % (Cin * n);
  int c = rem / n, s = rem % n;
  grouped[i] = (c < 3) ? xyz[((size_t)b * n + s) * 3 + c]
                       : feats[((size_t)b * Cf + (c - 3)) * n + s];
}

// ---------------------------------------------------------- WMMA fragments
// A = weights (O x C row major). 16-bit A 16x32 layout:
// lane m=lane&15, kb = lane<16 ? 0 : 8; elems 0..7 -> K=kb+0..7,
// elems 8..15 -> K=kb+16..23. Contiguous K -> immediate-offset loads.
__device__ __forceinline__ v16bf frag_a_fast(const float* __restrict__ W, int C_,
                                             int o0, int k0, int lane) {
  int m = lane & 15;
  int kb = (lane < 16) ? 0 : 8;
  const float* p = W + (size_t)(o0 + m) * C_ + k0 + kb;
  v16bf a;
#pragma unroll
  for (int j = 0; j < 8; ++j) a[j] = (__bf16)p[j];
#pragma unroll
  for (int j = 0; j < 8; ++j) a[8 + j] = (__bf16)p[16 + j];
  return a;
}

__device__ __forceinline__ v16bf frag_a_guard(const float* __restrict__ W, int O_,
                                              int C_, int o0, int k0, int lane) {
  int m = lane & 15;
  int kb = (lane < 16) ? 0 : 8;
  int o = o0 + m;
  bool ov = (o < O_);
  const float* Wrow = W + (size_t)(ov ? o : 0) * C_;
  v16bf a;
#pragma unroll
  for (int j = 0; j < 16; ++j) {
    int k = k0 + kb + ((j < 8) ? j : (8 + j));  // j>=8 -> kb+16+(j-8)
    float val = (ov && k < C_) ? Wrow[k] : 0.f;
    a[j] = (__bf16)val;
  }
  return a;
}

// B fragment from the LDS-staged 32(K) x 16(N) fp32 tile.
// lane n=lane&15, kb = lane<16 ? 0 : 8; elems 0..7 -> K = kb + j, elems 8..15 -> K = kb + 16 + (j-8).
__device__ __forceinline__ v16bf frag_b_lds(const float* __restrict__ sB,
                                            int lane) {
  int n = lane & 15;
  int kb = (lane < 16) ? 0 : 8;
  const float* p = sB + kb * 16 + n;  // stride 16 floats = 64B imm offsets
  v16bf bm;
#pragma unroll
  for (int j = 0; j < 16; ++j) bm[j] = (__bf16)p[((j < 8) ? j : (8 + j)) * 16];
  return bm;
}

__device__ __forceinline__ v16bf frag_b_guard(const float* __restrict__ Xb, int C_,
                                              int M_, int k0, int n0, int lane) {
  int n = n0 + (lane & 15);
  int kb = (lane < 16) ? 0 : 8;
  bool nv = (n < M_);
  v16bf bm;
#pragma unroll
  for (int j = 0; j < 16; ++j) {
    int k = k0 + kb + ((j < 8) ? j : (8 + j));
    float val = (nv && k < C_) ? Xb[(size_t)k * M_ + n] : 0.f;
    bm[j] = (__bf16)val;
  }
  return bm;
}

#define WMMA_BF16(A, Bf, Cacc) \
  __builtin_amdgcn_wmma_f32_16x16x32_bf16(false, (A), false, (Bf), (short)0, \
                                          (Cacc), false, false)

// mode 0: y=relu(acc*gamma/sqrt(1+eps)+beta), store Y[b,O,M]
// mode 1: y=acc+bias(g),                      store Y[b,O,M]
// mode 2: y=acc+bias(g),                      store Y[b,M,O] (classifier)
// mode 3: y=relu(bn),  atomicMax-pool over groups of nsample -> Y[b,O,M/ns]
__global__ void __launch_bounds__(64)
k_gemm_wmma(const float* __restrict__ W, const float* __restrict__ g,
            const float* __restrict__ bet, const float* __restrict__ X,
            float* __restrict__ Y, int C_, int O_, int M_, int mode,
            int nsample) {
  __shared__ float sB[32 * 16];        // one 32K x 16N fp32 activation tile
  int lane = threadIdx.x & 31;
  int wave = threadIdx.x >> 5;         // 2 waves per block
  int n0 = blockIdx.x * 16;
  int o0 = blockIdx.y * 64;            // 4 stacked 16-row o tiles (2 per wave)
  int o0w = o0 + wave * 32;
  int b = blockIdx.z;
  const float* Xb = X + (size_t)b * C_ * M_;
  v8f acc0 = {}, acc1 = {};
  int kfull = C_ & ~31;
  bool interior = (o0 + 64 <= O_) && (n0 + 16 <= M_);  // uniform per block
  if (interior) {
    // staging coords for this thread: row = t/2 (K), half-row = t&1 (8 floats)
    int srow = threadIdx.x >> 1;
    int scol = (threadIdx.x & 1) * 8;
    float* lp = sB + srow * 16 + scol;
    for (int k0 = 0; k0 < kfull; k0 += 32) {
      const float* gp = Xb + (size_t)(k0 + srow) * M_ + n0 + scol;
#if HAVE_ASYNC_LDS
      __builtin_amdgcn_global_load_async_to_lds_b128(
          (__attribute__((address_space(1))) v4i*)gp,
          (__attribute__((address_space(3))) v4i*)lp, 0, 0);
      __builtin_amdgcn_global_load_async_to_lds_b128(
          (__attribute__((address_space(1))) v4i*)(gp + 4),
          (__attribute__((address_space(3))) v4i*)(lp + 4), 0, 0);
      __builtin_amdgcn_s_wait_asynccnt(0);
#else
      *(float4*)lp = *(const float4*)gp;            // 16B aligned: M%16==0
      *(float4*)(lp + 4) = *(const float4*)(gp + 4);
#endif
      __syncthreads();                 // tile visible to both waves
      v16bf bb = frag_b_lds(sB, lane);
      v16bf a0 = frag_a_fast(W, C_, o0w, k0, lane);
      v16bf a1 = frag_a_fast(W, C_, o0w + 16, k0, lane);
      acc0 = WMMA_BF16(a0, bb, acc0);
      acc1 = WMMA_BF16(a1, bb, acc1);
      __syncthreads();                 // before next overwrite
    }
    if (kfull < C_) {
      v16bf bb = frag_b_guard(Xb, C_, M_, kfull, n0, lane);
      v16bf a0 = frag_a_guard(W, O_, C_, o0w, kfull, lane);
      v16bf a1 = frag_a_guard(W, O_, C_, o0w + 16, kfull, lane);
      acc0 = WMMA_BF16(a0, bb, acc0);
      acc1 = WMMA_BF16(a1, bb, acc1);
    }
  } else {
    for (int k0 = 0; k0 < C_; k0 += 32) {
      v16bf bb = frag_b_guard(Xb, C_, M_, k0, n0, lane);
      v16bf a0 = frag_a_guard(W, O_, C_, o0w, k0, lane);
      v16bf a1 = frag_a_guard(W, O_, C_, o0w + 16, k0, lane);
      acc0 = WMMA_BF16(a0, bb, acc0);
      acc1 = WMMA_BF16(a1, bb, acc1);
    }
  }
  int n = n0 + (lane & 15);
  int mbase = (lane < 16) ? 0 : 8;
#pragma unroll
  for (int t = 0; t < 2; ++t) {
    v8f acc = t ? acc1 : acc0;
#pragma unroll
    for (int r = 0; r < 8; ++r) {
      int o = o0w + t * 16 + mbase + r;
      if (o >= O_ || n >= M_) continue;
      float yv = acc[r];
      if (mode == 0 || mode == 3) {
        float gs = g[o] * rsqrtf(1.f + BN_EPS);
        yv = fmaxf(yv * gs + bet[o], 0.f);
      } else {
        yv += g[o];
      }
      if (mode == 0 || mode == 1) {
        Y[((size_t)b * O_ + o) * M_ + n] = yv;
      } else if (mode == 2) {
        Y[((size_t)b * M_ + n) * O_ + o] = yv;
      } else {
        int p = n / nsample;
        atomicMax((unsigned int*)&Y[((size_t)b * O_ + o) * (M_ / nsample) + p],
                  __float_as_uint(yv));  // yv >= 0 after ReLU
      }
    }
  }
}

// ------------------------------------------------- feature propagation bits
__global__ void k_knn3(const float* __restrict__ up, const float* __restrict__ dn,
                       int* __restrict__ idx3, float* __restrict__ w3, int B_,
                       int nu, int nd) {
  int i = blockIdx.x * blockDim.x + threadIdx.x;
  if (i >= B_ * nu) return;
  int b = i / nu;
  float px = up[(size_t)i * 3], py = up[(size_t)i * 3 + 1], pz = up[(size_t)i * 3 + 2];
  const float* D = dn + (size_t)b * nd * 3;
  float d0 = 1e30f, d1 = 1e30f, d2 = 1e30f;
  int i0 = 0, i1 = 0, i2 = 0;
  for (int j = 0; j < nd; ++j) {
    float dx = D[j * 3] - px, dy = D[j * 3 + 1] - py, dz = D[j * 3 + 2] - pz;
    float d = dx * dx + dy * dy + dz * dz;
    if (d < d0)      { d2 = d1; i2 = i1; d1 = d0; i1 = i0; d0 = d; i0 = j; }
    else if (d < d1) { d2 = d1; i2 = i1; d1 = d; i1 = j; }
    else if (d < d2) { d2 = d; i2 = j; }
  }
  float w0 = 1.f / fmaxf(d0, 1e-10f);
  float w1 = 1.f / fmaxf(d1, 1e-10f);
  float w2 = 1.f / fmaxf(d2, 1e-10f);
  float ws = w0 + w1 + w2;
  idx3[(size_t)i * 3] = i0; idx3[(size_t)i * 3 + 1] = i1; idx3[(size_t)i * 3 + 2] = i2;
  w3[(size_t)i * 3] = w0 / ws; w3[(size_t)i * 3 + 1] = w1 / ws; w3[(size_t)i * 3 + 2] = w2 / ws;
}

// comb (B, Cd+Cu, nu) = concat(3NN-interp(fd), fu)
__global__ void k_interp_concat(const float* __restrict__ fd,
                                const int* __restrict__ idx3,
                                const float* __restrict__ w3,
                                const float* __restrict__ fu,
                                float* __restrict__ comb, int B_, int Cd, int Cu,
                                int nu, int nd) {
  int Cin = Cd + Cu;
  int i = blockIdx.x * blockDim.x + threadIdx.x;
  if (i >= B_ * Cin * nu) return;
  int b = i / (Cin * nu);
  int rem = i % (Cin * nu);
  int c = rem / nu, n = rem % nu;
  float v;
  if (c < Cd) {
    const float* F = fd + ((size_t)b * Cd + c) * nd;
    size_t base = ((size_t)b * nu + n) * 3;
    v = F[idx3[base]] * w3[base] + F[idx3[base + 1]] * w3[base + 1] +
        F[idx3[base + 2]] * w3[base + 2];
  } else {
    v = fu[((size_t)b * Cu + (c - Cd)) * nu + n];
  }
  comb[i] = v;
}

// FP4: broadcast the single global feature, concat with f3
__global__ void k_bcast_concat(const float* __restrict__ fd /*B,Cd,1*/,
                               const float* __restrict__ fu /*B,Cu,n*/,
                               float* __restrict__ comb, int B_, int Cd, int Cu,
                               int n) {
  int Cin = Cd + Cu;
  int i = blockIdx.x * blockDim.x + threadIdx.x;
  if (i >= B_ * Cin * n) return;
  int b = i / (Cin * n);
  int rem = i % (Cin * n);
  int c = rem / n, s = rem % n;
  comb[i] = (c < Cd) ? fd[(size_t)b * Cd + c]
                     : fu[((size_t)b * Cu + (c - Cd)) * n + s];
}

// ------------------------------------------------------------------- host
extern "C" void kernel_launch(void* const* d_in, const int* in_sizes, int n_in,
                              void* d_out, int out_size, void* d_ws, size_t ws_size,
                              hipStream_t stream) {
  (void)in_sizes; (void)n_in; (void)out_size; (void)ws_size;
  const int Bc = 16, Nc = 4096;
  const float* x = (const float*)d_in[0];
  int pi = 1;
  struct L { const float *W, *g, *b; int ci, co; };
  auto nl = [&](int ci, int co) {
    L l{(const float*)d_in[pi], (const float*)d_in[pi + 1],
        (const float*)d_in[pi + 2], ci, co};
    pi += 3;
    return l;
  };
  // params in setup_inputs() insertion order: sa1..sa4, fp4..fp1, head, cls
  L sa1[3] = {nl(9, 64),    nl(64, 64),   nl(64, 128)};
  L sa2[3] = {nl(131, 128), nl(128, 128), nl(128, 256)};
  L sa3[3] = {nl(259, 256), nl(256, 256), nl(256, 512)};
  L sa4[3] = {nl(515, 512), nl(512, 512), nl(512, 1024)};
  L fp4[2] = {nl(1536, 512), nl(512, 512)};
  L fp3[2] = {nl(768, 256),  nl(256, 256)};
  L fp2[2] = {nl(384, 128),  nl(128, 128)};
  L fp1[2] = {nl(134, 128),  nl(128, 128)};
  L headL  = nl(128, 128);
  const float* Wc = (const float*)d_in[pi++];
  const float* bc = (const float*)d_in[pi++];

  // bump allocator over d_ws (~380 MB total)
  char* wsp = (char*)d_ws;
  auto alloc = [&](size_t bytes) -> void* {
    void* p = (void*)wsp;
    wsp += (bytes + 255) & ~(size_t)255;
    return p;
  };
  float* xyz0 = (float*)alloc((size_t)Bc * Nc * 3 * 4);
  float* feat0 = (float*)alloc((size_t)Bc * 6 * Nc * 4);
  float* xyz1 = (float*)alloc((size_t)Bc * 1024 * 3 * 4);
  float* f1 = (float*)alloc((size_t)Bc * 128 * 1024 * 4);
  float* xyz2 = (float*)alloc((size_t)Bc * 256 * 3 * 4);
  float* f2 = (float*)alloc((size_t)Bc * 256 * 256 * 4);
  float* xyz3 = (float*)alloc((size_t)Bc * 64 * 3 * 4);
  float* f3 = (float*)alloc((size_t)Bc * 512 * 64 * 4);
  float* f4 = (float*)alloc((size_t)Bc * 1024 * 4);
  float* up3 = (float*)alloc((size_t)Bc * 512 * 64 * 4);
  float* up2 = (float*)alloc((size_t)Bc * 256 * 256 * 4);
  float* up1 = (float*)alloc((size_t)Bc * 128 * 1024 * 4);
  float* up0 = (float*)alloc((size_t)Bc * 128 * 4096 * 4);
  float* hb = (float*)alloc((size_t)Bc * 128 * 4096 * 4);
  int* fidx = (int*)alloc((size_t)Bc * 1024 * 4);
  int* gidx = (int*)alloc((size_t)Bc * 1024 * 32 * 4);
  int* kidx = (int*)alloc((size_t)Bc * 4096 * 3 * 4);
  float* kw = (float*)alloc((size_t)Bc * 4096 * 3 * 4);
  const size_t AR = 34340864;  // max grouped/intermediate elt count
  float* arG = (float*)alloc(AR * 4);
  float* arA = (float*)alloc(AR * 4);

  auto cdiv = [](int a, int b) { return (a + b - 1) / b; };
  auto gemm = [&](const L& l, const float* Xp, float* Yp, int M, int mode, int ns) {
    dim3 gr(cdiv(M, 16), cdiv(l.co, 64), Bc);
    k_gemm_wmma<<<gr, 64, 0, stream>>>(l.W, l.g, l.b, Xp, Yp, l.ci, l.co, M,
                                       mode, ns);
  };
  auto fill0 = [&](float* p, int n) {
    k_fill_zero<<<cdiv(n, 256), 256, 0, stream>>>(p, n);
  };

  // input split
  k_split<<<cdiv(Bc * Nc, 256), 256, 0, stream>>>(x, xyz0, feat0, Bc, Nc, 9);

  // ---- set abstraction stages (grouped in arG -> arA -> arG -> pooled out)
  auto sa = [&](const float* xyzIn, const float* featsIn, int Nin, int np,
                float rad, int ns, L* Ls, int Cf, float* xyzOut, float* fOut) {
    size_t shmem = (size_t)(Nin + 256) * 4 + 256 * 4;
    k_fps<<<Bc, 256, shmem, stream>>>(xyzIn, fidx, Nin, np);
    k_gather_xyz<<<cdiv(Bc * np, 256), 256, 0, stream>>>(xyzIn, fidx, xyzOut,
                                                         Bc, Nin, np);
    k_ballquery<<<cdiv(Bc * np, 128), 128, 0, stream>>>(
        xyzIn, xyzOut, gidx, Bc, Nin, np, ns, rad * rad);
    int M = np * ns;
    k_group<<<cdiv(Bc * M, 256), 256, 0, stream>>>(xyzIn, xyzOut, featsIn, gidx,
                                                   arG, Bc, Nin, np, ns, Cf);
    gemm(Ls[0], arG, arA, M, 0, 0);
    gemm(Ls[1], arA, arG, M, 0, 0);
    fill0(fOut, Bc * Ls[2].co * np);
    gemm(Ls[2], arG, fOut, M, 3, ns);  // fused ReLU + max-pool
  };
  sa(xyz0, feat0, Nc, 1024, 0.5f, 32, sa1, 6, xyz1, f1);
  sa(xyz1, f1, 1024, 256, 1.0f, 64, sa2, 128, xyz2, f2);
  sa(xyz2, f2, 256, 64, 2.0f, 128, sa3, 256, xyz3, f3);

  // ---- SA4 (global)
  {
    int n = 64, Cf = 512, Cin = 515;
    k_sa4_group<<<cdiv(Bc * Cin * n, 256), 256, 0, stream>>>(xyz3, f3, arG, Bc,
                                                             n, Cf);
    gemm(sa4[0], arG, arA, n, 0, 0);
    gemm(sa4[1], arA, arG, n, 0, 0);
    fill0(f4, Bc * 1024);
    gemm(sa4[2], arG, f4, n, 3, n);  // pool over all 64
  }

  // ---- FP4: broadcast f4, concat f3 -> up3
  {
    int n = 64, Cd = 1024, Cu = 512, Cin = Cd + Cu;
    k_bcast_concat<<<cdiv(Bc * Cin * n, 256), 256, 0, stream>>>(f4, f3, arG, Bc,
                                                                Cd, Cu, n);
    gemm(fp4[0], arG, arA, n, 0, 0);
    gemm(fp4[1], arA, up3, n, 0, 0);
  }
  // ---- FP3: interp up3 (64->256), concat f2 -> up2
  {
    int nu = 256, nd = 64, Cd = 512, Cu = 256, Cin = Cd + Cu;
    k_knn3<<<cdiv(Bc * nu, 256), 256, 0, stream>>>(xyz2, xyz3, kidx, kw, Bc, nu, nd);
    k_interp_concat<<<cdiv(Bc * Cin * nu, 256), 256, 0, stream>>>(
        up3, kidx, kw, f2, arG, Bc, Cd, Cu, nu, nd);
    gemm(fp3[0], arG, arA, nu, 0, 0);
    gemm(fp3[1], arA, up2, nu, 0, 0);
  }
  // ---- FP2: interp up2 (256->1024), concat f1 -> up1
  {
    int nu = 1024, nd = 256, Cd = 256, Cu = 128, Cin = Cd + Cu;
    k_knn3<<<cdiv(Bc * nu, 256), 256, 0, stream>>>(xyz1, xyz2, kidx, kw, Bc, nu, nd);
    k_interp_concat<<<cdiv(Bc * Cin * nu, 256), 256, 0, stream>>>(
        up2, kidx, kw, f1, arG, Bc, Cd, Cu, nu, nd);
    gemm(fp2[0], arG, arA, nu, 0, 0);
    gemm(fp2[1], arA, up1, nu, 0, 0);
  }
  // ---- FP1: interp up1 (1024->4096), concat feat0 -> up0
  {
    int nu = 4096, nd = 1024, Cd = 128, Cu = 6, Cin = Cd + Cu;
    k_knn3<<<cdiv(Bc * nu, 256), 256, 0, stream>>>(xyz0, xyz1, kidx, kw, Bc, nu, nd);
    k_interp_concat<<<cdiv(Bc * Cin * nu, 256), 256, 0, stream>>>(
        up1, kidx, kw, feat0, arG, Bc, Cd, Cu, nu, nd);
    gemm(fp1[0], arG, arA, nu, 0, 0);
    gemm(fp1[1], arA, up0, nu, 0, 0);
  }

  // ---- head + classifier (writes d_out as (B, N, 6))
  gemm(headL, up0, hb, Nc, 0, 0);
  L clsL{Wc, bc, nullptr, 128, 6};
  gemm(clsL, hb, (float*)d_out, Nc, 2, 0);
}
